// DocREModel_52432960749969
// MI455X (gfx1250) — matrix operand order repaired and checked by reading, not compiled, not run on hardware
//
#include <hip/hip_runtime.h>

// ---------------- problem constants ----------------
constexpr int cB  = 4;
constexpr int cC  = 1024;
constexpr int cD  = 768;
constexpr int cH  = 12;
constexpr int cE  = 24;
constexpr int cM  = 8;
constexpr int cP  = 552;              // E*(E-1)
constexpr int cNC = 97;               // classes
constexpr int cK  = 12;               // D / BLOCK
constexpr int cBL = 64;               // BLOCK
constexpr int cD3 = 3 * cD;           // 2304
constexpr int cN  = cB * cP;          // 2208 pair rows
constexpr float NEG_INF = -1e30f;

typedef __attribute__((ext_vector_type(16))) __bf16 v16bf;
typedef __attribute__((ext_vector_type(8)))  __bf16 v8bf;
typedef __attribute__((ext_vector_type(8)))  float  v8f;
typedef int v4i __attribute__((vector_size(16)));
typedef __attribute__((address_space(1))) v4i* gv4i_p;   // global int4*
typedef __attribute__((address_space(3))) v4i* lv4i_p;   // LDS int4*

__device__ __forceinline__ v16bf cat8(v8bf lo, v8bf hi) {
    return __builtin_shufflevector(lo, hi, 0,1,2,3,4,5,6,7,8,9,10,11,12,13,14,15);
}

// ---------------- conversion kernels ----------------
__global__ void k_cvt_seqT(const float* __restrict__ seq, __bf16* __restrict__ seqT) {
    size_t i = (size_t)blockIdx.x * blockDim.x + threadIdx.x;
    if (i >= (size_t)cB * cD * cC) return;
    int c = (int)(i % cC);
    size_t t = i / cC;
    int d = (int)(t % cD);
    int b = (int)(t / cD);
    seqT[i] = (__bf16)seq[((size_t)b * cC + c) * cD + d];
}

__global__ void k_cvt_WT(const float* __restrict__ W, __bf16* __restrict__ WT) {
    size_t i = (size_t)blockIdx.x * blockDim.x + threadIdx.x;
    if (i >= (size_t)cD * cD3) return;
    int k = (int)(i % cD3);
    int n = (int)(i / cD3);
    WT[i] = (__bf16)W[(size_t)k * cD + n];
}

__global__ void k_cvt_flat(const float* __restrict__ src, __bf16* __restrict__ dst, size_t n) {
    size_t i = (size_t)blockIdx.x * blockDim.x + threadIdx.x;
    if (i < n) dst[i] = (__bf16)src[i];
}

// ---------------- entity pooling ----------------
__global__ void k_eemb(const float* __restrict__ seq, const float* __restrict__ mmask,
                       const int* __restrict__ midx, float* __restrict__ e_emb) {
    int be = blockIdx.x;
    int b = be / cE;
    const int*   idx = midx  + (size_t)be * cM;
    const float* msk = mmask + (size_t)be * cM;
    int rows[cM]; float mk[cM];
#pragma unroll
    for (int m = 0; m < cM; ++m) { rows[m] = idx[m] + 1; mk[m] = msk[m]; }
    for (int d = threadIdx.x; d < cD; d += blockDim.x) {
        float v[cM]; float mx = NEG_INF;
#pragma unroll
        for (int m = 0; m < cM; ++m) {
            float x = (mk[m] > 0.f) ? seq[((size_t)b * cC + rows[m]) * cD + d] : NEG_INF;
            v[m] = x; mx = fmaxf(mx, x);
        }
        float s = 0.f;
#pragma unroll
        for (int m = 0; m < cM; ++m) s += expf(v[m] - mx);
        e_emb[(size_t)be * cD + d] = mx + logf(s);
    }
}

__global__ void k_eatt(const float* __restrict__ att, const float* __restrict__ mmask,
                       const int* __restrict__ midx, float* __restrict__ e_att) {
    int beh = blockIdx.x;
    int h  = beh % cH;
    int be = beh / cH;
    int b  = be / cE;
    const int*   idx = midx  + (size_t)be * cM;
    const float* msk = mmask + (size_t)be * cM;
    int rows[cM]; float mk[cM]; float cnt = 0.f;
#pragma unroll
    for (int m = 0; m < cM; ++m) { rows[m] = idx[m] + 1; mk[m] = msk[m]; cnt += mk[m]; }
    cnt = fmaxf(cnt, 1.f);
    float inv = 1.f / cnt;
    for (int c = threadIdx.x; c < cC; c += blockDim.x) {
        float s = 0.f;
#pragma unroll
        for (int m = 0; m < cM; ++m)
            s += mk[m] * att[(((size_t)b * cH + h) * cC + rows[m]) * cC + c];
        e_att[((size_t)be * cH + h) * cC + c] = s * inv;
    }
}

// ---------------- pair features + normalized pair attention ----------------
__global__ void k_pair(const float* __restrict__ e_emb, const float* __restrict__ e_att,
                       const float* __restrict__ ner, const int* __restrict__ pairs,
                       const int* __restrict__ types,
                       __bf16* __restrict__ hs_fea, __bf16* __restrict__ ts_fea,
                       __bf16* __restrict__ htA) {
    int n = blockIdx.x;
    int b = n / cP;
    int he  = pairs[(size_t)n * 2 + 0];
    int te  = pairs[(size_t)n * 2 + 1];
    int hty = types[(size_t)n * 2 + 0];
    int tty = types[(size_t)n * 2 + 1];

    const float* hrow = e_emb + ((size_t)b * cE + he) * cD;
    const float* trow = e_emb + ((size_t)b * cE + te) * cD;
    for (int d = threadIdx.x; d < cD; d += blockDim.x) {
        hs_fea[(size_t)n * cD3 + d]          = (__bf16)hrow[d];
        ts_fea[(size_t)n * cD3 + d]          = (__bf16)trow[d];
        hs_fea[(size_t)n * cD3 + 2 * cD + d] = (__bf16)ner[(size_t)hty * cD + d];
        ts_fea[(size_t)n * cD3 + 2 * cD + d] = (__bf16)ner[(size_t)tty * cD + d];
    }

    const float* ha = e_att + ((size_t)b * cE + he) * cH * cC;
    const float* ta = e_att + ((size_t)b * cE + te) * cH * cC;
    float sv[4]; float loc = 0.f;
#pragma unroll
    for (int q = 0; q < 4; ++q) {
        int c = threadIdx.x + q * 256;
        float s = 0.f;
#pragma unroll
        for (int h = 0; h < cH; ++h) s += ha[h * cC + c] * ta[h * cC + c];
        s *= (1.f / cH);
        sv[q] = s; loc += s;
    }
    __shared__ float red[256];
    red[threadIdx.x] = loc;
    __syncthreads();
    for (int st = 128; st > 0; st >>= 1) {
        if (threadIdx.x < st) red[threadIdx.x] += red[threadIdx.x + st];
        __syncthreads();
    }
    float inv = 1.f / (red[0] + 1e-5f);
#pragma unroll
    for (int q = 0; q < 4; ++q) {
        int c = threadIdx.x + q * 256;
        htA[(size_t)n * cC + c] = (__bf16)(sv[q] * inv);
    }
}

// ---------------- WMMA GEMM 1: rs = ht_att @ seq (per batch) ----------------
__global__ void k_gemm_rs(const __bf16* __restrict__ htA, const __bf16* __restrict__ seqT,
                          __bf16* __restrict__ hs_fea, __bf16* __restrict__ ts_fea) {
    const int MT = 35, NT = 48;
    int wave = blockIdx.x * (blockDim.x >> 5) + (threadIdx.x >> 5);
    if (wave >= cB * MT * NT) return;
    int b   = wave / (MT * NT);
    int rem = wave % (MT * NT);
    int m0 = (rem / NT) * 16;
    int n0 = (rem % NT) * 16;
    int lane = threadIdx.x & 31, laneLo = lane & 15, laneHi = lane >> 4;

    int arow = m0 + laneLo; if (arow >= cP) arow = cP - 1;
    const __bf16* ap = htA  + ((size_t)b * cP + arow) * cC + laneHi * 8;
    const __bf16* bp = seqT + ((size_t)b * cD + n0 + laneLo) * cC + laneHi * 8;

    v8f acc = {};
    for (int kk = 0; kk < cC; kk += 32) {
        v16bf a  = cat8(*(const v8bf*)(ap + kk), *(const v8bf*)(ap + kk + 16));
        v16bf bb = cat8(*(const v8bf*)(bp + kk), *(const v8bf*)(bp + kk + 16));
        acc = __builtin_amdgcn_wmma_f32_16x16x32_bf16(false, a, false, bb,
                                                      (short)0, acc, false, false);
    }
#pragma unroll
    for (int v = 0; v < 8; ++v) {
        int row = m0 + v + laneHi * 8;
        if (row < cP) {
            int col = n0 + laneLo;
            size_t nidx = (size_t)b * cP + row;
            __bf16 val = (__bf16)acc[v];
            hs_fea[nidx * cD3 + cD + col] = val;
            ts_fea[nidx * cD3 + cD + col] = val;
        }
    }
}

// ---------------- WMMA GEMM 2: z = tanh(fea @ W + bias) ----------------
// Block = 4 waves sharing one n-tile; B panel async-staged into LDS in 768-K chunks.
constexpr int ZKC = 768;               // K chunk held in LDS
constexpr int ZLD = ZKC + 8;           // padded row stride (bf16) to dodge bank conflicts

__global__ void __launch_bounds__(128) k_gemm_z(const __bf16* __restrict__ fea,
                                                const __bf16* __restrict__ WT,
                                                const float* __restrict__ bias,
                                                float* __restrict__ z) {
    __shared__ __align__(16) __bf16 sB[16 * ZLD];    // ~24.3 KB

    const int NT = 48;
    int nt = blockIdx.x % NT;
    int mg = blockIdx.x / NT;                        // 0..34
    int wv = threadIdx.x >> 5;
    int mt = mg * 4 + wv;
    if (mt > (cN / 16) - 1) mt = (cN / 16) - 1;      // clamp: keep wave alive for barriers
    int m0 = mt * 16, n0 = nt * 16;
    int lane = threadIdx.x & 31, laneLo = lane & 15, laneHi = lane >> 4;

    const __bf16* ap = fea + (size_t)(m0 + laneLo) * cD3 + laneHi * 8;

    v8f acc = {};
    for (int kc = 0; kc < cD3; kc += ZKC) {
        // ---- stage B panel chunk (16 rows x ZKC bf16) into LDS ----
        {
            const int CHUNKS_PER_ROW = (ZKC * 2) / 16;           // 96 x 16B per row
            const int TOTAL16 = 16 * CHUNKS_PER_ROW;             // 1536 chunks
            for (int idx16 = threadIdx.x; idx16 < TOTAL16; idx16 += 128) {
                int row = idx16 / CHUNKS_PER_ROW;
                int c16 = idx16 % CHUNKS_PER_ROW;
                const __bf16* g = WT + (size_t)(n0 + row) * cD3 + kc + c16 * 8;
                __bf16* l = sB + row * ZLD + c16 * 8;
#if __has_builtin(__builtin_amdgcn_global_load_async_to_lds_b128)
                __builtin_amdgcn_global_load_async_to_lds_b128(
                    (gv4i_p)g, (lv4i_p)l, 0, 0);
#else
                *(v8bf*)l = *(const v8bf*)g;
#endif
            }
#if __has_builtin(__builtin_amdgcn_global_load_async_to_lds_b128)
#if __has_builtin(__builtin_amdgcn_s_wait_asynccnt)
            __builtin_amdgcn_s_wait_asynccnt(0);
#else
            asm volatile("s_wait_asynccnt 0" ::: "memory");
#endif
#endif
            __syncthreads();
        }
        // ---- WMMA over the staged chunk (B fragments from LDS) ----
        const __bf16* bp = sB + laneLo * ZLD + laneHi * 8;
        for (int kk = 0; kk < ZKC; kk += 32) {
            v16bf a  = cat8(*(const v8bf*)(ap + kc + kk), *(const v8bf*)(ap + kc + kk + 16));
            v16bf bb = cat8(*(const v8bf*)(bp + kk), *(const v8bf*)(bp + kk + 16));
            acc = __builtin_amdgcn_wmma_f32_16x16x32_bf16(false, a, false, bb,
                                                          (short)0, acc, false, false);
        }
        __syncthreads();
    }

    int col = n0 + laneLo;
    float bv = bias[col];
#pragma unroll
    for (int v = 0; v < 8; ++v) {
        int row = m0 + v + laneHi * 8;
        z[(size_t)row * cD + col] = tanhf(acc[v] + bv);
    }
}

// ---------------- WMMA GEMM 3: block bilinear classifier ----------------
__global__ void k_bilinear(const float* __restrict__ zh, const float* __restrict__ zt,
                           const __bf16* __restrict__ wbl, const float* __restrict__ bbl,
                           float* __restrict__ out) {
    const int CT = 7;
    int wave = blockIdx.x * (blockDim.x >> 5) + (threadIdx.x >> 5);
    if (wave >= (cN / 16) * CT) return;
    int m0 = (wave / CT) * 16;
    int r0 = (wave % CT) * 16;
    int lane = threadIdx.x & 31, laneLo = lane & 15, laneHi = lane >> 4;

    int row = m0 + laneLo;
    int rr  = r0 + laneLo;
    int rcl = (rr < cNC) ? rr : (cNC - 1);
    const float* zhr = zh + (size_t)row * cD;
    const float* ztr = zt + (size_t)row * cD;

    v8f acc = {};
    for (int k = 0; k < cK; ++k) {
        const float*  ztk = ztr + k * cBL;
        const __bf16* wk  = wbl + ((size_t)rcl * cK + k) * (cBL * cBL);
        if (k + 1 < cK)                      // prefetch next k's W block for this class
            __builtin_prefetch((const void*)(wk + cBL * cBL), 0, 0);
#pragma unroll
        for (int jc = 0; jc < 2; ++jc) {
            int j0 = jc * 32;
            const float* tp = ztk + j0 + laneHi * 8;
            float tl[8], th[8];
#pragma unroll
            for (int e = 0; e < 8; ++e) { tl[e] = tp[e]; th[e] = tp[16 + e]; }
            const __bf16* wp = wk + j0 + laneHi * 8;
            for (int i = 0; i < cBL; ++i) {
                float s = zhr[k * cBL + i];
                v16bf a;
#pragma unroll
                for (int e = 0; e < 8; ++e) {
                    a[e]     = (__bf16)(s * tl[e]);
                    a[8 + e] = (__bf16)(s * th[e]);
                }
                const __bf16* wpi = wp + i * cBL;
                v16bf bb = cat8(*(const v8bf*)(wpi), *(const v8bf*)(wpi + 16));
                acc = __builtin_amdgcn_wmma_f32_16x16x32_bf16(false, a, false, bb,
                                                              (short)0, acc, false, false);
            }
        }
    }
    if (rr < cNC) {
        float bv = bbl[rr];
#pragma unroll
        for (int v = 0; v < 8; ++v) {
            int m = m0 + v + laneHi * 8;
            out[(size_t)m * cNC + rr] = acc[v] + bv;
        }
    }
}

// ---------------- launcher ----------------
extern "C" void kernel_launch(void* const* d_in, const int* in_sizes, int n_in,
                              void* d_out, int out_size, void* d_ws, size_t ws_size,
                              hipStream_t stream) {
    const float* seq   = (const float*)d_in[0];
    const float* att   = (const float*)d_in[1];
    const float* mmask = (const float*)d_in[2];
    const float* ner   = (const float*)d_in[3];
    const float* Wh    = (const float*)d_in[4];
    const float* bh    = (const float*)d_in[5];
    const float* Wt    = (const float*)d_in[6];
    const float* bt    = (const float*)d_in[7];
    const float* Wbl   = (const float*)d_in[8];
    const float* bbl   = (const float*)d_in[9];
    const int*   midx  = (const int*)d_in[10];
    const int*   pairs = (const int*)d_in[11];
    const int*   types = (const int*)d_in[12];
    float* out = (float*)d_out;

    char* ws = (char*)d_ws;
    size_t off = 0;
    auto take = [&](size_t bytes) -> char* {
        char* p = ws + off;
        off += (bytes + 255) & ~(size_t)255;
        return p;
    };
    float*  e_emb  = (float*)  take((size_t)cB * cE * cD * 4);
    float*  e_att  = (float*)  take((size_t)cB * cE * cH * cC * 4);
    __bf16* htA    = (__bf16*) take((size_t)cB * cP * cC * 2);
    __bf16* seqT   = (__bf16*) take((size_t)cB * cD * cC * 2);
    __bf16* hs_fea = (__bf16*) take((size_t)cN * cD3 * 2);
    __bf16* ts_fea = (__bf16*) take((size_t)cN * cD3 * 2);
    __bf16* WhT    = (__bf16*) take((size_t)cD * cD3 * 2);
    __bf16* WtT    = (__bf16*) take((size_t)cD * cD3 * 2);
    float*  zh     = (float*)  take((size_t)cN * cD * 4);
    float*  zt     = (float*)  take((size_t)cN * cD * 4);
    __bf16* wblb   = (__bf16*) take((size_t)cNC * cK * cBL * cBL * 2);

    {
        size_t n = (size_t)cB * cD * cC;
        k_cvt_seqT<<<(unsigned)((n + 255) / 256), 256, 0, stream>>>(seq, seqT);
    }
    {
        size_t n = (size_t)cD * cD3;
        k_cvt_WT<<<(unsigned)((n + 255) / 256), 256, 0, stream>>>(Wh, WhT);
        k_cvt_WT<<<(unsigned)((n + 255) / 256), 256, 0, stream>>>(Wt, WtT);
    }
    {
        size_t n = (size_t)cNC * cK * cBL * cBL;
        k_cvt_flat<<<(unsigned)((n + 255) / 256), 256, 0, stream>>>(Wbl, wblb, n);
    }

    k_eemb<<<cB * cE, 256, 0, stream>>>(seq, mmask, midx, e_emb);
    k_eatt<<<cB * cE * cH, 256, 0, stream>>>(att, mmask, midx, e_att);

    k_pair<<<cN, 256, 0, stream>>>(e_emb, e_att, ner, pairs, types, hs_fea, ts_fea, htA);

    k_gemm_rs<<<(cB * 35 * 48) / 4, 128, 0, stream>>>(htA, seqT, hs_fea, ts_fea);

    // 48 n-tiles x 35 m-groups (4 m-tiles per block, clamped)
    k_gemm_z<<<48 * 35, 128, 0, stream>>>(hs_fea, WhT, bh, zh);
    k_gemm_z<<<48 * 35, 128, 0, stream>>>(ts_fea, WtT, bt, zt);

    {
        int tiles = (cN / 16) * 7;
        int blocks = (tiles + 3) / 4;
        k_bilinear<<<blocks, 128, 0, stream>>>(zh, zt, wblb, bbl, out);
    }
    (void)in_sizes; (void)n_in; (void)out_size; (void)ws_size;
}
